// LGL_INR_5196910428706
// MI455X (gfx1250) — matrix operand order repaired and checked by reading, not compiled, not run on hardware
//
#include <hip/hip_runtime.h>
#include <hip/hip_bf16.h>

typedef __attribute__((ext_vector_type(16))) _Float16 v16h;
typedef __attribute__((ext_vector_type(8)))  float    v8f;

#define CLS   128          // C
#define KCH   32           // samples per K-chunk (one WMMA K-step = 32)
#define EPSF  1e-7f
#define CHUNK_BYTES (KCH * CLS * 4)   // 16384

// ws layout (floats):
//   [0      .. 16383]  sum_probs   (row-major: class j * 128 + col k)
//   [16384  .. 32767]  sum_logneg
//   [32768  .. 32895]  counts
//   [32896  .. 33023]  poslog (sum of log(p_true+eps) per class)
#define WS_FLOATS 33024

__global__ void zero_ws_kernel(float* ws, int n) {
    int i = blockIdx.x * blockDim.x + threadIdx.x;
    if (i < n) ws[i] = 0.0f;
}

static __device__ __forceinline__ v16h load_btile(const _Float16* s, int colBase) {
    v16h b;
#pragma unroll
    for (int v = 0; v < 8; ++v) {
        int o = (v < 4) ? 2 * v : 16 + 2 * (v - 4);
        b[2 * v]     = s[colBase + o];
        b[2 * v + 1] = s[colBase + o + 1];
    }
    return b;
}

__global__ __launch_bounds__(256)
void segsum_kernel(const float* __restrict__ in, const int* __restrict__ tgt,
                   float* __restrict__ ws, int numChunks) {
    // Double-buffered raw chunk (async DMA destination), 2 x 16KB
    __shared__ __align__(16) float    sRaw[2][KCH * CLS];
    // Transformed f16 tiles, stored TRANSPOSED: s[col*KCH + row]
    __shared__ __align__(16) _Float16 sP[CLS * KCH];   // probs f16
    __shared__ __align__(16) _Float16 sL[CLS * KCH];   // log(1-p+eps) f16
    __shared__ int      sT[KCH];         // targets of this chunk
    __shared__ float    sCnt[CLS];       // per-class count (WG partial)
    __shared__ float    sPos[CLS];       // per-class sum log(p_true+eps)

    const int tid  = threadIdx.x;        // 0..255
    const int wave = tid >> 5;           // 0..7 (wave32)
    const int lane = tid & 31;
    const int kh   = lane >> 4;          // lane half: 0 or 1
    const int mrow = lane & 15;          // M row within tile / N col within tile

    if (tid < CLS) { sCnt[tid] = 0.0f; sPos[tid] = 0.0f; }

    const int mtile   = wave;            // wave owns classes [16*mtile, 16*mtile+16)
    const int myClass = mtile * 16 + mrow;

    v8f accP[8], accL[8];
    v8f vzero = {};
#pragma unroll
    for (int nt = 0; nt < 8; ++nt) { accP[nt] = vzero; accL[nt] = vzero; }

    // --- async DMA issue: copy one 16KB chunk into sRaw[b] (4 x b128 per lane) ---
    auto issue_async = [&](int chunk, int b) {
        unsigned gbyte = (unsigned)chunk * (unsigned)CHUNK_BYTES + (unsigned)tid * 16u;
        unsigned lbyte = (unsigned)(uintptr_t)(&sRaw[b][0]) + (unsigned)tid * 16u;
#pragma unroll
        for (int j = 0; j < 4; ++j) {
            asm volatile("global_load_async_to_lds_b128 %0, %1, %2"
                         :: "v"(lbyte + (unsigned)j * 4096u),
                            "v"(gbyte + (unsigned)j * 4096u),
                            "s"(in)
                         : "memory");
        }
    };

    int buf = 0;
    if (blockIdx.x < (unsigned)numChunks) issue_async(blockIdx.x, buf);

    for (int ch = blockIdx.x; ch < numChunks; ch += gridDim.x) {
        const int base    = ch * KCH;
        const int nch     = ch + (int)gridDim.x;
        const bool hasNext = (nch < numChunks);

        // Stream next chunk behind this one's transform + WMMA, then wait for
        // the current chunk (ASYNCcnt completes in order: 4 left == next batch).
        if (hasNext) {
            issue_async(nch, buf ^ 1);
            asm volatile("s_wait_asynccnt 0x4" ::: "memory");
        } else {
            asm volatile("s_wait_asynccnt 0x0" ::: "memory");
        }
        __syncthreads();  // raw[buf] visible WG-wide; prev iteration's LDS readers done

        if (tid < KCH) {
            int t = tgt[base + tid];
            sT[tid] = t;
            atomicAdd(&sCnt[t], 1.0f);   // ds_add_f32
        }
        __syncthreads();

        // Transform 32x128 elements: sigmoid + log(1-p+eps), packed f16, transposed
#pragma unroll
        for (int i = 0; i < 16; ++i) {
            int e = tid + 256 * i;       // 0..4095
            int r = e >> 7;              // sample row 0..31
            int c = e & 127;             // class col 0..127
            float x  = sRaw[buf][e];
            float p  = 1.0f / (1.0f + __expf(-x));
            float ln = __logf(1.0f - p + EPSF);
            sP[c * KCH + r] = (_Float16)p;
            sL[c * KCH + r] = (_Float16)ln;
            if (c == sT[r]) atomicAdd(&sPos[c], __logf(p + EPSF));  // ds_add_f32
        }
        __syncthreads();

        // One-hot A tile (16x32 f16, MxK) per ISA layout:
        //   lanes 0-15: M=lane; VGPR v<4: K = kh*8+2v,2v+1; v>=4: K = 16+kh*8+2(v-4),+1
        v16h a;
#pragma unroll
        for (int v = 0; v < 4; ++v) {
            int K0 = kh * 8 + 2 * v;
            a[2 * v]     = (sT[K0]     == myClass) ? (_Float16)1.0f : (_Float16)0.0f;
            a[2 * v + 1] = (sT[K0 + 1] == myClass) ? (_Float16)1.0f : (_Float16)0.0f;
        }
#pragma unroll
        for (int v = 4; v < 8; ++v) {
            int K0 = 16 + kh * 8 + 2 * (v - 4);
            a[2 * v]     = (sT[K0]     == myClass) ? (_Float16)1.0f : (_Float16)0.0f;
            a[2 * v + 1] = (sT[K0 + 1] == myClass) ? (_Float16)1.0f : (_Float16)0.0f;
        }

        // B tile (32x16 f16, KxN): VGPR v<4: K = kh*8+2v,+1; v>=4: K = 16+kh*8+2(v-4),+1 (N=lane&15).
        // Transposed LDS => each pair is one b32.
        // Software-pipelined: issue tile nt+1 loads before tile nt's WMMAs.
        auto cb = [&](int nt) { return (nt * 16 + mrow) * KCH + kh * 8; };
        v16h bp_c = load_btile(sP, cb(0));
        v16h bl_c = load_btile(sL, cb(0));
#pragma unroll
        for (int nt = 0; nt < 8; ++nt) {
            v16h bp_n, bl_n;
            if (nt < 7) {
                bp_n = load_btile(sP, cb(nt + 1));
                bl_n = load_btile(sL, cb(nt + 1));
            }
            accP[nt] = __builtin_amdgcn_wmma_f32_16x16x32_f16(
                false, a, false, bp_c, (short)0, accP[nt], false, false);
            accL[nt] = __builtin_amdgcn_wmma_f32_16x16x32_f16(
                false, a, false, bl_c, (short)0, accL[nt], false, false);
            if (nt < 7) { bp_c = bp_n; bl_c = bl_n; }
        }

        buf ^= 1;
    }

    __syncthreads();

    // Flush: C/D layout — VGPR r: lanes0-15 -> M=r, lanes16-31 -> M=r+8; N = lane&15
    float* wsP = ws;
    float* wsL = ws + 16384;
    float* wsC = ws + 32768;
    float* wsG = ws + 32896;
#pragma unroll
    for (int nt = 0; nt < 8; ++nt) {
#pragma unroll
        for (int r = 0; r < 8; ++r) {
            int row = mtile * 16 + r + 8 * kh;
            int col = nt * 16 + mrow;
            atomicAdd(&wsP[row * CLS + col], accP[nt][r]);
            atomicAdd(&wsL[row * CLS + col], accL[nt][r]);
        }
    }
    if (tid < CLS) {
        atomicAdd(&wsC[tid], sCnt[tid]);
        atomicAdd(&wsG[tid], sPos[tid]);
    }
}

__global__ __launch_bounds__(128)
void finalize_kernel(const float* __restrict__ ws, float* __restrict__ out) {
    __shared__ float sCnt[CLS];
    __shared__ float red[CLS];
    const int k = threadIdx.x;          // column owned by this thread
    const float* wsP = ws;
    const float* wsL = ws + 16384;
    const float* wsC = ws + 32768;
    const float* wsG = ws + 32896;

    float cnt = wsC[k];
    sCnt[k] = cnt;
    __syncthreads();

    // positive term for class k
    float pos = (cnt > 0.0f) ? (wsG[k] / cnt) : 0.0f;

    // column-k softmax over present rows j != k (online), weights * mean_loss_neg
    float m = -1e30f, s = 0.0f, w = 0.0f;
    for (int j = 0; j < CLS; ++j) {
        float cj = sCnt[j];
        if (j == k || !(cj > 0.0f)) continue;
        float logit = wsP[j * CLS + k] / cj;        // BETA = 1
        float mln   = wsL[j * CLS + k] / cj;
        if (logit > m) {
            float sc = __expf(m - logit);
            s *= sc; w *= sc; m = logit;
        }
        float e = __expf(logit - m);
        s += e; w += e * mln;
    }
    float neg = (s > 0.0f) ? (w / s) : 0.0f;

    red[k] = pos + neg;
    __syncthreads();
#pragma unroll
    for (int off = 64; off > 0; off >>= 1) {
        if (k < off) red[k] += red[k + off];
        __syncthreads();
    }
    if (k == 0) out[0] = -red[0];
}

extern "C" void kernel_launch(void* const* d_in, const int* in_sizes, int n_in,
                              void* d_out, int out_size, void* d_ws, size_t ws_size,
                              hipStream_t stream) {
    const float* in  = (const float*)d_in[0];
    const int*   tg  = (const int*)d_in[1];
    float*       ws  = (float*)d_ws;
    float*       out = (float*)d_out;

    const int B = in_sizes[0] / CLS;          // 262144
    const int numChunks = B / KCH;            // 8192

    zero_ws_kernel<<<(WS_FLOATS + 255) / 256, 256, 0, stream>>>(ws, WS_FLOATS);
    segsum_kernel<<<512, 256, 0, stream>>>(in, tg, ws, numChunks);
    finalize_kernel<<<1, 128, 0, stream>>>(ws, out);
}
